// Q_GoBERT_83511344103730
// MI455X (gfx1250) — matrix operand order repaired and checked by reading, not compiled, not run on hardware
//
#include <hip/hip_runtime.h>
#include <math.h>

// Problem constants (fixed by the reference)
#define T_STEPS 50
#define QD      32
#define CH      128   // H*C1 == C2 == 128
#define NH      4
#define C1V     32
#define LHV     128
#define EPSV    1e-5f

typedef __attribute__((ext_vector_type(16))) _Float16 v16h;
typedef __attribute__((ext_vector_type(8)))  float    v8f;

// ---------- helpers ----------
__device__ __forceinline__ unsigned encf(float f) {
  unsigned u = __float_as_uint(f);
  return (u & 0x80000000u) ? ~u : (u | 0x80000000u);
}
__device__ __forceinline__ float decf(unsigned u) {
  unsigned v = (u & 0x80000000u) ? (u & 0x7FFFFFFFu) : ~u;
  return __uint_as_float(v);
}
__device__ __forceinline__ float lreluf(float v) { return v > 0.f ? v : 0.2f * v; }
__device__ __forceinline__ float eluf(float v)   { return v > 0.f ? v : (expf(v) - 1.f); }
__device__ __forceinline__ float sigf(float v)   { return 1.f / (1.f + expf(-v)); }

// ---------- fill kernels ----------
__global__ void k_fill_f32(float* p, float v, long n) {
  long i = blockIdx.x * (long)blockDim.x + threadIdx.x;
  if (i < n) p[i] = v;
}
__global__ void k_fill_enc(unsigned* p, float v, long n) {
  long i = blockIdx.x * (long)blockDim.x + threadIdx.x;
  if (i < n) p[i] = encf(v);
}

// ---------- B pre-pack: f32 weights -> f16 fragment-major ----------
// B element (k,n) at Bm[k*bks + n*bns].  Packed layout: for 16-col tile tn,
// K-step ks (32 wide), lane (16g+col), 16 contiguous f16 (K = ks*32 + 16g + i).
__global__ void k_pack_b(const float* __restrict__ Bm, int bks, int bns,
                         _Float16* __restrict__ Bp, int Nc, int K) {
  long idx = blockIdx.x * (long)blockDim.x + threadIdx.x;
  if (idx >= (long)Nc * K) return;
  const int ksteps = K >> 5;
  int i    = (int)(idx & 15);
  int lane = (int)((idx >> 4) & 31);
  long rest = idx >> 9;
  int ks   = (int)(rest % ksteps);
  int tn   = (int)(rest / ksteps);
  int g = lane >> 4, col16 = lane & 15;
  int k = ks * 32 + g * 16 + i;
  int n = tn * 16 + col16;
  Bp[idx] = (_Float16)Bm[(size_t)k * bks + (size_t)n * bns];
}

// ---------- WMMA GEMM: C = A[M,K] * B[K,Nc] (+ Cin) ----------
// A row-major f32 (lda), B pre-packed f16 (see k_pack_b), Cout row-major (ldc).
// One wave computes a 16x64 strip (4 accumulators, A fragment reused 4x).
// KSTEPS > 0: compile-time K/32 (full unroll, immediate-offset B loads).
// All per-lane offsets are 32-bit (buffers < 2^31 elements).
// Requires M%16==0, Nc%64==0, K%32==0 (guaranteed by this workload).
template <int KSTEPS>
__global__ void k_wmma_gemm(const float* __restrict__ A, int lda,
                            const _Float16* __restrict__ Bp,
                            const float* __restrict__ Cin, int ldcin,
                            float* __restrict__ Cout, int ldc,
                            int M, int Nc, int K) {
  const int lane = threadIdx.x & 31;
  const int wave = threadIdx.x >> 5;
  const int tilesN4 = Nc >> 6;                      // strips of 4 N-tiles
  const long tile = (long)blockIdx.x * (blockDim.x >> 5) + wave;
  const long total = (long)(M >> 4) * tilesN4;
  if (tile >= total) return;
  const int tm  = (int)(tile / tilesN4);
  const int tn4 = (int)(tile % tilesN4);
  const int g = lane >> 4;                          // lane half
  const int l = lane & 15;
  const int rowA = tm * 16 + l;
  const int ksteps = (KSTEPS > 0) ? KSTEPS : (K >> 5);

  const unsigned rowC  = (unsigned)(tm * 16 + g * 8);      // first C row this lane
  const unsigned colB0 = (unsigned)(tn4 * 64 + l);         // first C col this lane

  v8f acc[4];
  if (Cin) {
#pragma unroll
    for (int j = 0; j < 4; ++j) {
      const unsigned coff = rowC * (unsigned)ldcin + colB0 + (unsigned)(j * 16);
#pragma unroll
      for (int r = 0; r < 8; ++r)
        acc[j][r] = Cin[coff + (unsigned)(r * ldcin)];
    }
  } else {
#pragma unroll
    for (int j = 0; j < 4; ++j)
#pragma unroll
      for (int r = 0; r < 8; ++r) acc[j][r] = 0.f;
  }

  const float* arow = A + (unsigned)rowA * (unsigned)lda;
  __builtin_prefetch(arow, 0, 3);                   // global_prefetch_b8
  const _Float16* bbase = Bp + ((unsigned)(tn4 * 4 * ksteps) * 32u + (unsigned)lane) * 16u;

#pragma unroll
  for (int ks = 0; ks < ksteps; ++ks) {
    const float* ar = arow + ks * 32;
    // A fragment: one contiguous run of 16 floats (k = 16g + s) -> 4 x b128 + packed cvt
    const float4* p0 = (const float4*)(ar + g * 16);
    const float4* p1 = (const float4*)(ar + g * 16 + 8);
    float4 a00 = p0[0], a01 = p0[1], a10 = p1[0], a11 = p1[1];
    v16h a;
    a[0]  = (_Float16)a00.x; a[1]  = (_Float16)a00.y;
    a[2]  = (_Float16)a00.z; a[3]  = (_Float16)a00.w;
    a[4]  = (_Float16)a01.x; a[5]  = (_Float16)a01.y;
    a[6]  = (_Float16)a01.z; a[7]  = (_Float16)a01.w;
    a[8]  = (_Float16)a10.x; a[9]  = (_Float16)a10.y;
    a[10] = (_Float16)a10.z; a[11] = (_Float16)a10.w;
    a[12] = (_Float16)a11.x; a[13] = (_Float16)a11.y;
    a[14] = (_Float16)a11.z; a[15] = (_Float16)a11.w;
    // preload all 4 B fragments (immediate offsets when KSTEPS is constant)
    v16h b0 = *(const v16h*)(bbase + (unsigned)((0 * ksteps + ks) * 512));
    v16h b1 = *(const v16h*)(bbase + (unsigned)((1 * ksteps + ks) * 512));
    v16h b2 = *(const v16h*)(bbase + (unsigned)((2 * ksteps + ks) * 512));
    v16h b3 = *(const v16h*)(bbase + (unsigned)((3 * ksteps + ks) * 512));
    acc[0] = __builtin_amdgcn_wmma_f32_16x16x32_f16(false, a, false, b0,
                                                    (short)0, acc[0], false, false);
    acc[1] = __builtin_amdgcn_wmma_f32_16x16x32_f16(false, a, false, b1,
                                                    (short)0, acc[1], false, false);
    acc[2] = __builtin_amdgcn_wmma_f32_16x16x32_f16(false, a, false, b2,
                                                    (short)0, acc[2], false, false);
    acc[3] = __builtin_amdgcn_wmma_f32_16x16x32_f16(false, a, false, b3,
                                                    (short)0, acc[3], false, false);
  }

#pragma unroll
  for (int j = 0; j < 4; ++j) {
    const unsigned coff = rowC * (unsigned)ldc + colB0 + (unsigned)(j * 16);
#pragma unroll
    for (int r = 0; r < 8; ++r)
      Cout[coff + (unsigned)(r * ldc)] = acc[j][r];
  }
}

// ---------- GAT attention coefficients ----------
__global__ void k_attn(const float* __restrict__ xw,
                       const float* __restrict__ att_src,
                       const float* __restrict__ att_dst,
                       float* __restrict__ a_src, float* __restrict__ a_dst, long N) {
  long i = blockIdx.x * (long)blockDim.x + threadIdx.x;
  if (i >= N * NH) return;
  long node = i >> 2; int h = (int)(i & 3);
  const float4* v  = (const float4*)(xw + node * CH + h * C1V);
  const float4* ws = (const float4*)(att_src + h * C1V);
  const float4* wd = (const float4*)(att_dst + h * C1V);
  float as = 0.f, ad = 0.f;
#pragma unroll
  for (int q = 0; q < C1V / 4; ++q) {
    float4 t = v[q], s = ws[q], d = wd[q];
    as += t.x * s.x + t.y * s.y + t.z * s.z + t.w * s.w;
    ad += t.x * d.x + t.y * d.y + t.z * d.z + t.w * d.w;
  }
  a_src[i] = as; a_dst[i] = ad;
}

__device__ __forceinline__ void edge_sd(const int* ei, long E, long ed, int& s, int& d) {
  if (ed < E) { s = ei[ed]; d = ei[E + ed]; }
  else        { s = d = (int)(ed - E); }    // self-loop
}

// pass 1: per-dst max (encoded) + degree
__global__ void k_gat_max_deg(const int* __restrict__ ei, long E, long N,
                              const float* __restrict__ a_src,
                              const float* __restrict__ a_dst,
                              unsigned* __restrict__ mbuf, float* __restrict__ deg) {
  long idx = blockIdx.x * (long)blockDim.x + threadIdx.x;
  if (idx >= E + N) return;
  int s, d; edge_sd(ei, E, idx, s, d);
  atomicAdd(&deg[d], 1.0f);
#pragma unroll
  for (int h = 0; h < NH; ++h) {
    float e = lreluf(a_src[(size_t)s * NH + h] + a_dst[(size_t)d * NH + h]);
    atomicMax(&mbuf[(size_t)d * NH + h], encf(e));
  }
}

// pass 2: sum of exp(e - m)
__global__ void k_gat_sum(const int* __restrict__ ei, long E, long N,
                          const float* __restrict__ a_src,
                          const float* __restrict__ a_dst,
                          const unsigned* __restrict__ mbuf, float* __restrict__ sbuf) {
  long idx = blockIdx.x * (long)blockDim.x + threadIdx.x;
  if (idx >= (E + N) * NH) return;
  long ed = idx >> 2; int h = (int)(idx & 3);
  int s, d; edge_sd(ei, E, ed, s, d);
  float e = lreluf(a_src[(size_t)s * NH + h] + a_dst[(size_t)d * NH + h]);
  atomicAdd(&sbuf[(size_t)d * NH + h], expf(e - decf(mbuf[(size_t)d * NH + h])));
}

// pass 3: scatter alpha * xw[src] into hacc[dst]
__global__ void k_gat_msg(const int* __restrict__ ei, long E, long N,
                          const float* __restrict__ a_src,
                          const float* __restrict__ a_dst,
                          const unsigned* __restrict__ mbuf,
                          const float* __restrict__ sbuf,
                          const float* __restrict__ xw, float* __restrict__ hacc) {
  long idx = blockIdx.x * (long)blockDim.x + threadIdx.x;
  if (idx >= (E + N) * NH) return;
  long ed = idx >> 2; int h = (int)(idx & 3);
  int s, d; edge_sd(ei, E, ed, s, d);
  float e = lreluf(a_src[(size_t)s * NH + h] + a_dst[(size_t)d * NH + h]);
  float alpha = expf(e - decf(mbuf[(size_t)d * NH + h])) /
                (sbuf[(size_t)d * NH + h] + 1e-16f);
  const float4* xs = (const float4*)(xw + (size_t)s * CH + h * C1V);
  float vv[C1V];
#pragma unroll
  for (int q = 0; q < C1V / 4; ++q) {
    float4 t = xs[q];
    vv[4*q] = t.x; vv[4*q+1] = t.y; vv[4*q+2] = t.z; vv[4*q+3] = t.w;
  }
  float* hd = hacc + (size_t)d * CH + h * C1V;
#pragma unroll
  for (int cc = 0; cc < C1V; ++cc) atomicAdd(&hd[cc], vv[cc] * alpha);
}

// elementwise: h = bn(elu(h + bias)) in place, channels = CH
__global__ void k_bias_elu_bn(float* __restrict__ h, const float* __restrict__ bias,
                              const float* __restrict__ gg, const float* __restrict__ bb,
                              const float* __restrict__ mm, const float* __restrict__ vv,
                              long n) {
  long i = blockIdx.x * (long)blockDim.x + threadIdx.x;
  if (i >= n) return;
  int cc = (int)(i & (CH - 1));
  float v = eluf(h[i] + bias[cc]);
  h[i] = gg[cc] * (v - mm[cc]) * rsqrtf(vv[cc] + EPSV) + bb[cc];
}

__global__ void k_dinv(float* __restrict__ deg, long N) {
  long i = blockIdx.x * (long)blockDim.x + threadIdx.x;
  if (i >= N) return;
  float d = deg[i];
  deg[i] = d > 0.f ? rsqrtf(d) : 0.f;
}

// GCN normalized scatter: h2[dst] += hw[src] * dinv[src]*dinv[dst]
__global__ void k_gcn_edge(const int* __restrict__ ei, long E, long N,
                           const float* __restrict__ dinv,
                           const float* __restrict__ hw, float* __restrict__ h2) {
  long idx = blockIdx.x * (long)blockDim.x + threadIdx.x;
  if (idx >= (E + N) * 4) return;
  long ed = idx >> 2; int q = (int)(idx & 3);
  int s, d; edge_sd(ei, E, ed, s, d);
  float norm = dinv[s] * dinv[d];
  const float4* hs = (const float4*)(hw + (size_t)s * CH + q * 32);
  float vv[32];
#pragma unroll
  for (int p = 0; p < 8; ++p) {
    float4 t = hs[p];
    vv[4*p] = t.x; vv[4*p+1] = t.y; vv[4*p+2] = t.z; vv[4*p+3] = t.w;
  }
  float* hd = h2 + (size_t)d * CH + q * 32;
#pragma unroll
  for (int cc = 0; cc < 32; ++cc) atomicAdd(&hd[cc], vv[cc] * norm);
}

// gate[i] = h2[i] . gate_W + gate_b
__global__ void k_gate(const float* __restrict__ h2, const float* __restrict__ gW,
                       const float* __restrict__ gb, float* __restrict__ gate, long N) {
  long i = blockIdx.x * (long)blockDim.x + threadIdx.x;
  if (i >= N) return;
  const float4* v = (const float4*)(h2 + (size_t)i * CH);
  const float4* w = (const float4*)gW;
  float acc = gb[0];
#pragma unroll
  for (int q = 0; q < CH / 4; ++q) {
    float4 t = v[q], u = w[q];
    acc += t.x * u.x + t.y * u.y + t.z * u.z + t.w * u.w;
  }
  gate[i] = acc;
}

__global__ void k_pool_max(const int* __restrict__ batch, const float* __restrict__ gate,
                           unsigned* __restrict__ gm, long N) {
  long i = blockIdx.x * (long)blockDim.x + threadIdx.x;
  if (i >= N) return;
  atomicMax(&gm[batch[i]], encf(gate[i]));
}
__global__ void k_pool_sum(const int* __restrict__ batch, const float* __restrict__ gate,
                           const unsigned* __restrict__ gm, float* __restrict__ gs, long N) {
  long i = blockIdx.x * (long)blockDim.x + threadIdx.x;
  if (i >= N) return;
  int b = batch[i];
  atomicAdd(&gs[b], expf(gate[i] - decf(gm[b])));
}
__global__ void k_pool_acc(const int* __restrict__ batch, const float* __restrict__ gate,
                           const unsigned* __restrict__ gm, const float* __restrict__ gs,
                           const float* __restrict__ h2, float* __restrict__ grep, long N) {
  long idx = blockIdx.x * (long)blockDim.x + threadIdx.x;
  if (idx >= N * 4) return;
  long i = idx >> 2; int q = (int)(idx & 3);
  int b = batch[i];
  float score = expf(gate[i] - decf(gm[b])) / (gs[b] + 1e-16f);
  const float4* hs = (const float4*)(h2 + (size_t)i * CH + q * 32);
  float vv[32];
#pragma unroll
  for (int p = 0; p < 8; ++p) {
    float4 t = hs[p];
    vv[4*p] = t.x; vv[4*p+1] = t.y; vv[4*p+2] = t.z; vv[4*p+3] = t.w;
  }
  float* gd = grep + (size_t)b * CH + q * 32;
#pragma unroll
  for (int cc = 0; cc < 32; ++cc) atomicAdd(&gd[cc], score * vv[cc]);
}

// LSTM gate activations (torch order i,f,g,o)
__global__ void k_lstm_act(const float* __restrict__ gates, const float* __restrict__ bih,
                           const float* __restrict__ bhh, float* __restrict__ hb,
                           float* __restrict__ cb, int Bg) {
  int i = blockIdx.x * blockDim.x + threadIdx.x;
  if (i >= Bg * LHV) return;
  int b = i >> 7, j = i & (LHV - 1);
  const float* g = gates + (size_t)b * 4 * LHV;
  float ig = sigf(g[j]            + bih[j]            + bhh[j]);
  float fg = sigf(g[LHV + j]      + bih[LHV + j]      + bhh[LHV + j]);
  float gg = tanhf(g[2*LHV + j]   + bih[2*LHV + j]    + bhh[2*LHV + j]);
  float og = sigf(g[3*LHV + j]    + bih[3*LHV + j]    + bhh[3*LHV + j]);
  float c = fg * cb[i] + ig * gg;
  cb[i] = c;
  hb[i] = og * tanhf(c);
}

__global__ void k_final(const float* __restrict__ grep, const float* __restrict__ hb,
                        const float* __restrict__ fcW, const float* __restrict__ fcb,
                        float* __restrict__ out, int Bg) {
  int b = blockIdx.x * blockDim.x + threadIdx.x;
  if (b >= Bg) return;
  float acc = fcb[0];
#pragma unroll 8
  for (int cc = 0; cc < CH; ++cc)  acc += grep[(size_t)b * CH + cc] * fcW[cc];
#pragma unroll 8
  for (int cc = 0; cc < LHV; ++cc) acc += hb[(size_t)b * LHV + cc] * fcW[CH + cc];
  out[b] = acc;
}

// ---------- host orchestration ----------
static inline int nblk(long n, int t) { return (int)((n + t - 1) / t); }

extern "C" void kernel_launch(void* const* d_in, const int* in_sizes, int n_in,
                              void* d_out, int out_size, void* d_ws, size_t ws_size,
                              hipStream_t stream) {
  const float* x       = (const float*)d_in[0];
  const int*   ei      = (const int*)d_in[1];
  const int*   batch   = (const int*)d_in[2];
  const float* quant   = (const float*)d_in[3];
  const float* gat_W   = (const float*)d_in[4];
  const float* att_src = (const float*)d_in[5];
  const float* att_dst = (const float*)d_in[6];
  const float* gat_b   = (const float*)d_in[7];
  const float* bn1_g   = (const float*)d_in[8];
  const float* bn1_b   = (const float*)d_in[9];
  const float* bn1_m   = (const float*)d_in[10];
  const float* bn1_v   = (const float*)d_in[11];
  const float* gcn_W   = (const float*)d_in[12];
  const float* gcn_b   = (const float*)d_in[13];
  const float* bn2_g   = (const float*)d_in[14];
  const float* bn2_b   = (const float*)d_in[15];
  const float* bn2_m   = (const float*)d_in[16];
  const float* bn2_v   = (const float*)d_in[17];
  const float* gate_W  = (const float*)d_in[18];
  const float* gate_b  = (const float*)d_in[19];
  const float* Wih     = (const float*)d_in[20];
  const float* Whh     = (const float*)d_in[21];
  const float* bih     = (const float*)d_in[22];
  const float* bhh     = (const float*)d_in[23];
  const float* fc_W    = (const float*)d_in[24];
  const float* fc_b    = (const float*)d_in[25];
  float* out = (float*)d_out;

  const long N  = in_sizes[0] / CH;
  const long E  = in_sizes[1] / 2;
  const long Bg = in_sizes[3] / (T_STEPS * QD);
  const long EN = E + N;

  // ---- workspace carve (allocations rounded to 64B for vector loads) ----
  float* ws = (float*)d_ws;
  size_t off = 0;
  auto alloc = [&](size_t n) { n = (n + 15) & ~(size_t)15; float* p = ws + off; off += n; return p; };
  const size_t nch    = (size_t)N * CH;
  const size_t pre_sz = (size_t)Bg * T_STEPS * 4 * LHV;
  float*    slab0 = alloc(nch);                                 // xw, then h2
  float*    slab1 = alloc(nch > pre_sz ? nch : pre_sz);         // hacc/h, then precomp
  float*    hw    = alloc(nch);
  float*    a_src = alloc((size_t)N * NH);
  float*    a_dst = alloc((size_t)N * NH);
  unsigned* mbuf  = (unsigned*)alloc((size_t)N * NH);
  float*    sbuf  = alloc((size_t)N * NH);
  float*    deg   = alloc((size_t)N);                           // later dinv
  float*    gate  = alloc((size_t)N);
  unsigned* gm    = (unsigned*)alloc((size_t)Bg);
  float*    gs    = alloc((size_t)Bg);
  float*    grep  = alloc((size_t)Bg * CH);
  float*    gates = alloc((size_t)Bg * 4 * LHV);
  float*    hb    = alloc((size_t)Bg * LHV);
  float*    cb    = alloc((size_t)Bg * LHV);
  // packed f16 weight buffers (halves stored in float ws)
  _Float16* pGat = (_Float16*)alloc((size_t)CH * CH / 2);
  _Float16* pGcn = (_Float16*)alloc((size_t)CH * CH / 2);
  _Float16* pIh  = (_Float16*)alloc((size_t)QD * 4 * LHV / 2);
  _Float16* pHh  = (_Float16*)alloc((size_t)LHV * 4 * LHV / 2);
  (void)ws_size; (void)n_in; (void)out_size;

  auto gemm = [&](const float* A, int lda, const _Float16* Bp,
                  const float* Cin, int ldcin, float* Cout, int ldc,
                  long M, int Nc, int K) {
    long tiles = (M / 16) * (Nc / 64);          // one wave per 16x64 strip
    int blocks = nblk(tiles * 32, 256);
    if (K == 32)
      k_wmma_gemm<1><<<blocks, 256, 0, stream>>>(A, lda, Bp, Cin, ldcin, Cout, ldc,
                                                 (int)M, Nc, K);
    else if (K == 128)
      k_wmma_gemm<4><<<blocks, 256, 0, stream>>>(A, lda, Bp, Cin, ldcin, Cout, ldc,
                                                 (int)M, Nc, K);
    else
      k_wmma_gemm<0><<<blocks, 256, 0, stream>>>(A, lda, Bp, Cin, ldcin, Cout, ldc,
                                                 (int)M, Nc, K);
  };
  auto packb = [&](const float* Bm, int bks, int bns, _Float16* Bp, int Nc, int K) {
    k_pack_b<<<nblk((long)Nc * K, 256), 256, 0, stream>>>(Bm, bks, bns, Bp, Nc, K);
  };

  // ---- init accumulators + pack all weights ----
  k_fill_f32<<<nblk(nch, 256), 256, 0, stream>>>(slab1, 0.f, (long)nch);       // hacc
  k_fill_enc<<<nblk((long)N * NH, 256), 256, 0, stream>>>(mbuf, -1e30f, N * NH);
  k_fill_f32<<<nblk((long)N * NH, 256), 256, 0, stream>>>(sbuf, 0.f, N * NH);
  k_fill_f32<<<nblk(N, 256), 256, 0, stream>>>(deg, 0.f, N);
  packb(gat_W, CH, 1, pGat, CH, CH);
  packb(gcn_W, CH, 1, pGcn, CH, CH);
  packb(Wih, 1, QD, pIh, 4 * LHV, QD);
  packb(Whh, 1, LHV, pHh, 4 * LHV, LHV);

  // ---- GAT ----
  gemm(x, CH, pGat, nullptr, 0, slab0, CH, N, CH, CH);                         // xw
  k_attn<<<nblk(N * NH, 256), 256, 0, stream>>>(slab0, att_src, att_dst, a_src, a_dst, N);
  k_gat_max_deg<<<nblk(EN, 256), 256, 0, stream>>>(ei, E, N, a_src, a_dst, mbuf, deg);
  k_gat_sum<<<nblk(EN * NH, 256), 256, 0, stream>>>(ei, E, N, a_src, a_dst, mbuf, sbuf);
  k_gat_msg<<<nblk(EN * NH, 256), 256, 0, stream>>>(ei, E, N, a_src, a_dst, mbuf, sbuf,
                                                    slab0, slab1);
  k_bias_elu_bn<<<nblk(nch, 256), 256, 0, stream>>>(slab1, gat_b, bn1_g, bn1_b,
                                                    bn1_m, bn1_v, (long)nch);  // h

  // ---- GCN ----
  gemm(slab1, CH, pGcn, nullptr, 0, hw, CH, N, CH, CH);                        // hw
  k_dinv<<<nblk(N, 256), 256, 0, stream>>>(deg, N);
  k_fill_f32<<<nblk(nch, 256), 256, 0, stream>>>(slab0, 0.f, (long)nch);       // h2 accum
  k_gcn_edge<<<nblk(EN * 4, 256), 256, 0, stream>>>(ei, E, N, deg, hw, slab0);
  k_bias_elu_bn<<<nblk(nch, 256), 256, 0, stream>>>(slab0, gcn_b, bn2_g, bn2_b,
                                                    bn2_m, bn2_v, (long)nch);  // h2

  // ---- GlobalAttention pooling ----
  k_gate<<<nblk(N, 256), 256, 0, stream>>>(slab0, gate_W, gate_b, gate, N);
  k_fill_enc<<<nblk(Bg, 256), 256, 0, stream>>>(gm, -1e30f, Bg);
  k_fill_f32<<<nblk(Bg, 256), 256, 0, stream>>>(gs, 0.f, Bg);
  k_fill_f32<<<nblk((long)Bg * CH, 256), 256, 0, stream>>>(grep, 0.f, Bg * CH);
  k_pool_max<<<nblk(N, 256), 256, 0, stream>>>(batch, gate, gm, N);
  k_pool_sum<<<nblk(N, 256), 256, 0, stream>>>(batch, gate, gm, gs, N);
  k_pool_acc<<<nblk(N * 4, 256), 256, 0, stream>>>(batch, gate, gm, gs, slab0, grep, N);

  // ---- LSTM ----  (h is dead; slab1 becomes precomp = X @ Wih^T)
  gemm(quant, QD, pIh, nullptr, 0, slab1, 4 * LHV, Bg * T_STEPS, 4 * LHV, QD);
  k_fill_f32<<<nblk((long)Bg * LHV, 256), 256, 0, stream>>>(hb, 0.f, Bg * LHV);
  k_fill_f32<<<nblk((long)Bg * LHV, 256), 256, 0, stream>>>(cb, 0.f, Bg * LHV);
  for (int t = 0; t < T_STEPS; ++t) {
    // gates[b,:] = precomp[(b*T + t), :] + hb[b,:] @ Whh^T
    gemm(hb, LHV, pHh, slab1 + (size_t)t * 4 * LHV, T_STEPS * 4 * LHV,
         gates, 4 * LHV, Bg, 4 * LHV, LHV);
    k_lstm_act<<<nblk((long)Bg * LHV, 256), 256, 0, stream>>>(gates, bih, bhh, hb, cb,
                                                              (int)Bg);
  }

  // ---- final linear ----
  k_final<<<nblk(Bg, 256), 256, 0, stream>>>(grep, hb, fc_W, fc_b, out, (int)Bg);
}
